// GraphEncoder1_45930380264179
// MI455X (gfx1250) — matrix operand order repaired and checked by reading, not compiled, not run on hardware
//
#include <hip/hip_runtime.h>
#include <hip/hip_bf16.h>
#include <stdint.h>

typedef __attribute__((ext_vector_type(16))) _Float16 v16h;
typedef __attribute__((ext_vector_type(8)))  _Float16 v8h;
typedef __attribute__((ext_vector_type(8)))  float    v8f;

#define NN 200000
#define EE 3200000
#define BB 512
#define DD 32

// pre-packed B-fragment tile indices in wpack[16][32][16] (f16)
#define T_W0  0   // 2 tiles
#define T_WM  2   // 2 tiles
#define T_WIH 4   // 6 tiles (gate rows 0..95)
#define T_WHH 10  // 6 tiles

__device__ __forceinline__ float sigmoidf_(float x) { return 1.0f / (1.0f + __expf(-x)); }

__device__ __forceinline__ v16h load_a_frag_f32(const float* __restrict__ row, int kShA)
{
  // A-fragment halves j=0..7 -> k = kShA..kShA+7 ; j=8..15 -> k = 16+kShA..16+kShA+7
  const float4 p0 = *(const float4*)(row + kShA);
  const float4 p1 = *(const float4*)(row + kShA + 4);
  const float4 p2 = *(const float4*)(row + 16 + kShA);
  const float4 p3 = *(const float4*)(row + 16 + kShA + 4);
  v16h a;
  a[0] = (_Float16)p0.x; a[1] = (_Float16)p0.y; a[2] = (_Float16)p0.z; a[3] = (_Float16)p0.w;
  a[4] = (_Float16)p1.x; a[5] = (_Float16)p1.y; a[6] = (_Float16)p1.z; a[7] = (_Float16)p1.w;
  a[8] = (_Float16)p2.x; a[9] = (_Float16)p2.y; a[10] = (_Float16)p2.z; a[11] = (_Float16)p2.w;
  a[12] = (_Float16)p3.x; a[13] = (_Float16)p3.y; a[14] = (_Float16)p3.z; a[15] = (_Float16)p3.w;
  return a;
}

// ---------------------------------------------------------------------------
// Kernel 0: pack all weight B-fragments (f32 -> f16, WMMA B lane-major layout)
// wpack[T][lane][j] = M[((lane&15) + 16*tt) * 32 + (j&7) + 8*(lane>=16) + 16*(j>>3)]
// ---------------------------------------------------------------------------
__global__ void k_pack_w(const float* __restrict__ W0, const float* __restrict__ Wm,
                         const float* __restrict__ Wih, const float* __restrict__ Whh,
                         _Float16* __restrict__ wp)
{
  const int item = blockIdx.x * blockDim.x + threadIdx.x;   // 0..511
  if (item >= 16 * 32) return;
  const int T = item >> 5, l = item & 31;
  const float* M; int tt;
  if (T < T_WM)       { M = W0;  tt = T; }
  else if (T < T_WIH) { M = Wm;  tt = T - T_WM; }
  else if (T < T_WHH) { M = Wih; tt = T - T_WIH; }
  else                { M = Whh; tt = T - T_WHH; }
  const int row = (l & 15) + 16 * tt;
  const int kb  = (l >> 4) * 8;
  _Float16* d = wp + ((size_t)T * 32 + l) * 16;
#pragma unroll
  for (int j = 0; j < 16; ++j) d[j] = (_Float16)M[row * DD + kb + (j & 7) + 16 * (j >> 3)];
}

// ---------------------------------------------------------------------------
// Kernel 1: out = relu(emb[node_type] @ W0.T + b0)  (one 16-row tile per wave)
// ---------------------------------------------------------------------------
__global__ void __launch_bounds__(256) k_embed_lin0(
    const int* __restrict__ node_type, const float* __restrict__ emb,
    const _Float16* __restrict__ wp, const float* __restrict__ b0,
    float* __restrict__ h, int n)
{
  const int warp = threadIdx.x >> 5;
  const int lane = threadIdx.x & 31;
  const int wave = blockIdx.x * (blockDim.x >> 5) + warp;
  const int row0 = wave << 4;
  if (row0 >= n) return;                       // wave-uniform: EXEC all-1 for WMMA
  const int rl = lane & 15;
  const int hi = lane >> 4;
  const int kShA = hi ? 8 : 0;
  const v16h* wt = (const v16h*)wp;

  const float* xr = emb + (size_t)node_type[row0 + rl] * DD;
  const v16h a = load_a_frag_f32(xr, kShA);

#pragma unroll
  for (int nt = 0; nt < 2; ++nt) {
    const int col = rl + 16 * nt;
    const v16h b = wt[(T_W0 + nt) * 32 + lane];
    v8f c = {};
    c = __builtin_amdgcn_wmma_f32_16x16x32_f16(false, a, false, b, (short)0, c, false, false);
    const float bias = b0[col];
#pragma unroll
    for (int r = 0; r < 8; ++r) {
      const int rowg = row0 + r + (hi ? 8 : 0);
      const float v = c[r] + bias;
      h[(size_t)rowg * DD + col] = v > 0.f ? v : 0.f;
    }
  }
}

// ---------------------------------------------------------------------------
// Kernel 2: edge scatter  agg[dst] += out[src]   (bandwidth-bound hot loop)
// ---------------------------------------------------------------------------
__global__ void __launch_bounds__(256) k_scatter(
    const int* __restrict__ src, const int* __restrict__ dst,
    const float* __restrict__ hout, float* __restrict__ agg, int ne)
{
  const long long idx = (long long)blockIdx.x * blockDim.x + threadIdx.x;
  const int e = (int)(idx >> 3);
  if (e >= ne) return;
  const int c = ((int)idx & 7) * 4;
  const float4 v = *(const float4*)(hout + (size_t)src[e] * DD + c);
  float* d = agg + (size_t)dst[e] * DD + c;
  atomicAdd(d + 0, v.x); atomicAdd(d + 1, v.y);
  atomicAdd(d + 2, v.z); atomicAdd(d + 3, v.w);
}

// ---------------------------------------------------------------------------
// Kernel 3: fused  m = relu(agg @ Wm.T);  h = GRU(m, h)    (14 WMMAs / wave)
// ---------------------------------------------------------------------------
__global__ void __launch_bounds__(256) k_gru(
    const float* __restrict__ agg, float* __restrict__ h,
    const _Float16* __restrict__ wp,
    const float* __restrict__ bih, const float* __restrict__ bhh, int n)
{
  __shared__ _Float16 lm[8][16][DD];           // per-wave f16 staging of m tile
  const int warp = threadIdx.x >> 5;
  const int lane = threadIdx.x & 31;
  const int wave = blockIdx.x * (blockDim.x >> 5) + warp;
  const int row0 = wave << 4;
  const bool active = row0 < n;
  const int rl = lane & 15;
  const int hi = lane >> 4;
  const int kShA = hi ? 8 : 0;
  const v16h* wt = (const v16h*)wp;

  if (active) {
    const v16h a = load_a_frag_f32(agg + (size_t)(row0 + rl) * DD, kShA);
#pragma unroll
    for (int nt = 0; nt < 2; ++nt) {
      const int col = rl + 16 * nt;
      const v16h b = wt[(T_WM + nt) * 32 + lane];
      v8f c = {};
      c = __builtin_amdgcn_wmma_f32_16x16x32_f16(false, a, false, b, (short)0, c, false, false);
#pragma unroll
      for (int r = 0; r < 8; ++r) {
        const float v = c[r];
        lm[warp][r + (hi ? 8 : 0)][col] = (_Float16)(v > 0.f ? v : 0.f);
      }
    }
  }
  __syncthreads();
  if (!active) return;

  // A fragments: m from LDS (2 x ds_load_b128), h from global (f32 -> f16)
  const v8h m0 = *(const v8h*)&lm[warp][rl][kShA];
  const v8h m1 = *(const v8h*)&lm[warp][rl][16 + kShA];
  const v16h am = __builtin_shufflevector(m0, m1, 0, 1, 2, 3, 4, 5, 6, 7,
                                          8, 9, 10, 11, 12, 13, 14, 15);
  const v16h ah = load_a_frag_f32(h + (size_t)(row0 + rl) * DD, kShA);

#pragma unroll
  for (int t2 = 0; t2 < 2; ++t2) {
    const int col = rl + 16 * t2;              // gate-internal feature column 0..31
    v8f gir = {}, ghr = {}, giz = {}, ghz = {}, gin = {}, ghn = {};
    gir = __builtin_amdgcn_wmma_f32_16x16x32_f16(false, am, false,
          wt[(T_WIH + t2) * 32 + lane], (short)0, gir, false, false);
    ghr = __builtin_amdgcn_wmma_f32_16x16x32_f16(false, ah, false,
          wt[(T_WHH + t2) * 32 + lane], (short)0, ghr, false, false);
    giz = __builtin_amdgcn_wmma_f32_16x16x32_f16(false, am, false,
          wt[(T_WIH + 2 + t2) * 32 + lane], (short)0, giz, false, false);
    ghz = __builtin_amdgcn_wmma_f32_16x16x32_f16(false, ah, false,
          wt[(T_WHH + 2 + t2) * 32 + lane], (short)0, ghz, false, false);
    gin = __builtin_amdgcn_wmma_f32_16x16x32_f16(false, am, false,
          wt[(T_WIH + 4 + t2) * 32 + lane], (short)0, gin, false, false);
    ghn = __builtin_amdgcn_wmma_f32_16x16x32_f16(false, ah, false,
          wt[(T_WHH + 4 + t2) * 32 + lane], (short)0, ghn, false, false);

    const float bir  = bih[col],           bhr = bhh[col];
    const float biz  = bih[DD + col],      bhz = bhh[DD + col];
    const float bin_ = bih[2 * DD + col],  bhn = bhh[2 * DD + col];
#pragma unroll
    for (int r = 0; r < 8; ++r) {
      const int rowg = row0 + r + (hi ? 8 : 0);
      const float R  = sigmoidf_(gir[r] + bir + ghr[r] + bhr);
      const float Z  = sigmoidf_(giz[r] + biz + ghz[r] + bhz);
      const float Nv = tanhf(gin[r] + bin_ + R * (ghn[r] + bhn));
      float* hp = h + (size_t)rowg * DD + col;
      const float hv = *hp;
      *hp = (1.f - Z) * Nv + Z * hv;
    }
  }
}

// ---------------------------------------------------------------------------
// Kernel 4: segment start offsets from sorted graph_ids (handles empty graphs)
// ---------------------------------------------------------------------------
__global__ void k_segstart(const int* __restrict__ gid, int* __restrict__ start,
                           int n, int nb)
{
  const int i = blockIdx.x * blockDim.x + threadIdx.x;
  if (i >= n) return;
  const int g  = gid[i];
  const int gp = (i == 0) ? -1 : gid[i - 1];
  for (int b = gp + 1; b <= g; ++b) start[b] = i;
  if (i == n - 1)
    for (int b = g + 1; b <= nb; ++b) start[b] = n;
}

// ---------------------------------------------------------------------------
// Kernel 5: 3-layer LSTM + q (tiny: B=512). One block per graph.
// ---------------------------------------------------------------------------
__global__ void __launch_bounds__(128) k_lstm_q(
    float* __restrict__ q_star, float* __restrict__ q,
    float* __restrict__ hs, float* __restrict__ cs,
    const float* __restrict__ Wih0, const float* __restrict__ Whh0,
    const float* __restrict__ bih0, const float* __restrict__ bhh0,
    const float* __restrict__ Wih12, const float* __restrict__ Whh12,
    const float* __restrict__ bih12, const float* __restrict__ bhh12)
{
  const int b = blockIdx.x;
  const int t = threadIdx.x;                   // 0..127
  __shared__ float x[2 * DD];
  __shared__ float g[4 * DD];
  if (t < 2 * DD) x[t] = q_star[b * 2 * DD + t];
  __syncthreads();

  { // layer 0: input width 64
    float acc = bih0[t] + bhh0[t];
    const float* hrow = hs + (size_t)b * DD;
    for (int k = 0; k < 2 * DD; ++k) acc += x[k] * Wih0[t * 2 * DD + k];
    for (int k = 0; k < DD; ++k)     acc += hrow[k] * Whh0[t * DD + k];
    g[t] = acc;
    __syncthreads();
    if (t < DD) {
      const float i_ = g[t], f_ = g[DD + t], gg = g[2 * DD + t], o_ = g[3 * DD + t];
      const float c0 = cs[(size_t)b * DD + t];
      const float c2 = sigmoidf_(f_) * c0 + sigmoidf_(i_) * tanhf(gg);
      const float h2 = sigmoidf_(o_) * tanhf(c2);
      cs[(size_t)b * DD + t] = c2;
      hs[(size_t)b * DD + t] = h2;
      x[t] = h2;
    }
    __syncthreads();
  }
  for (int l = 0; l < 2; ++l) { // layers 1..2: input width 32
    const float* Wih = Wih12 + (size_t)l * 4 * DD * DD;
    const float* Whh = Whh12 + (size_t)l * 4 * DD * DD;
    float acc = bih12[l * 4 * DD + t] + bhh12[l * 4 * DD + t];
    const float* hrow = hs + (size_t)(l + 1) * BB * DD + (size_t)b * DD;
    for (int k = 0; k < DD; ++k) acc += x[k] * Wih[t * DD + k] + hrow[k] * Whh[t * DD + k];
    g[t] = acc;
    __syncthreads();
    if (t < DD) {
      const float i_ = g[t], f_ = g[DD + t], gg = g[2 * DD + t], o_ = g[3 * DD + t];
      float* cp = cs + (size_t)(l + 1) * BB * DD + (size_t)b * DD + t;
      float* hp = hs + (size_t)(l + 1) * BB * DD + (size_t)b * DD + t;
      const float c2 = sigmoidf_(f_) * (*cp) + sigmoidf_(i_) * tanhf(gg);
      const float h2 = sigmoidf_(o_) * tanhf(c2);
      *cp = c2; *hp = h2; x[t] = h2;
    }
    __syncthreads();
  }
  if (t < DD) {
    q[(size_t)b * DD + t] = x[t];
    q_star[(size_t)b * 2 * DD + t] = x[t];
  }
}

// ---------------------------------------------------------------------------
// Kernel 6: segment softmax attention + weighted readout r (one block/graph)
// ---------------------------------------------------------------------------
__global__ void __launch_bounds__(256) k_attn(
    const float* __restrict__ hout, const float* __restrict__ q,
    const int* __restrict__ start, float* __restrict__ e,
    float* __restrict__ q_star)
{
  const int b  = blockIdx.x;
  const int s0 = start[b], s1 = start[b + 1];
  const int lane = threadIdx.x & 31, warp = threadIdx.x >> 5;
  __shared__ float red[8];
  __shared__ float rred[8][DD];
  const float qv = q[(size_t)b * DD + lane];

  // phase 1: logits e_i = <out_i, q_b>, per-wave running max
  float mmax = -1e30f;
  for (int i = s0 + warp; i < s1; i += 8) {
    float v = hout[(size_t)i * DD + lane] * qv;
#pragma unroll
    for (int off = 16; off; off >>= 1) v += __shfl_xor(v, off, 32);
    if (lane == 0) e[i] = v;
    mmax = fmaxf(mmax, v);                     // v uniform after butterfly
  }
  if (lane == 0) red[warp] = mmax;
  __syncthreads();
  if (threadIdx.x == 0) {
    float m = red[0];
    for (int w = 1; w < 8; ++w) m = fmaxf(m, red[w]);
    red[0] = m;
  }
  __syncthreads();
  const float m = red[0];
  __syncthreads();

  // phase 2: p_i = exp(e_i - m) stored back into e; S = sum p
  float s = 0.f;
  for (int i = s0 + (int)threadIdx.x; i < s1; i += 256) {
    const float p = __expf(e[i] - m);
    e[i] = p;
    s += p;
  }
#pragma unroll
  for (int off = 16; off; off >>= 1) s += __shfl_xor(s, off, 32);
  if (lane == 0) red[warp] = s;
  __syncthreads();
  if (threadIdx.x == 0) {
    float S = 0.f;
    for (int w = 0; w < 8; ++w) S += red[w];
    red[0] = S;
  }
  __syncthreads();
  const float Sinv = (s1 > s0) ? 1.f / red[0] : 0.f;

  // phase 3: r = sum_i (p_i / S) * out_i
  float rloc = 0.f;
  for (int i = s0 + warp; i < s1; i += 8)
    rloc += e[i] * hout[(size_t)i * DD + lane];
  rred[warp][lane] = rloc * Sinv;
  __syncthreads();
  if (threadIdx.x < DD) {
    float r = 0.f;
    for (int w = 0; w < 8; ++w) r += rred[w][threadIdx.x];
    q_star[(size_t)b * 2 * DD + DD + threadIdx.x] = r;
  }
}

// ---------------------------------------------------------------------------
// Kernel 7: readout MLP  out = relu(q_star @ Wr1.T + br1) @ Wr2.T + br2
// ---------------------------------------------------------------------------
__global__ void __launch_bounds__(32) k_readout(
    const float* __restrict__ q_star,
    const float* __restrict__ Wr1, const float* __restrict__ br1,
    const float* __restrict__ Wr2, const float* __restrict__ br2,
    float* __restrict__ out)
{
  const int b = blockIdx.x, t = threadIdx.x;   // 32 threads
  __shared__ float x[2 * DD];
  __shared__ float hid[DD];
  x[t] = q_star[(size_t)b * 2 * DD + t];
  x[DD + t] = q_star[(size_t)b * 2 * DD + DD + t];
  __syncthreads();
  float acc = br1[t];
  for (int k = 0; k < 2 * DD; ++k) acc += x[k] * Wr1[t * 2 * DD + k];
  hid[t] = acc > 0.f ? acc : 0.f;
  __syncthreads();
  float acc2 = br2[t];
  for (int k = 0; k < DD; ++k) acc2 += hid[k] * Wr2[t * DD + k];
  out[(size_t)b * DD + t] = acc2;
}

// ---------------------------------------------------------------------------
extern "C" void kernel_launch(void* const* d_in, const int* in_sizes, int n_in,
                              void* d_out, int out_size, void* d_ws, size_t ws_size,
                              hipStream_t stream)
{
  const int*   node_type = (const int*)d_in[0];
  const int*   src       = (const int*)d_in[1];
  const int*   dst       = (const int*)d_in[2];
  const int*   gid       = (const int*)d_in[3];
  const float* emb   = (const float*)d_in[4];
  const float* W0    = (const float*)d_in[5];
  const float* b0    = (const float*)d_in[6];
  const float* Wm    = (const float*)d_in[7];
  const float* Wih_g = (const float*)d_in[8];
  const float* Whh_g = (const float*)d_in[9];
  const float* bih_g = (const float*)d_in[10];
  const float* bhh_g = (const float*)d_in[11];
  const float* Wih0  = (const float*)d_in[12];
  const float* Whh0  = (const float*)d_in[13];
  const float* bih0  = (const float*)d_in[14];
  const float* bhh0  = (const float*)d_in[15];
  const float* Wih12 = (const float*)d_in[16];
  const float* Whh12 = (const float*)d_in[17];
  const float* bih12 = (const float*)d_in[18];
  const float* bhh12 = (const float*)d_in[19];
  const float* Wr1   = (const float*)d_in[20];
  const float* br1   = (const float*)d_in[21];
  const float* Wr2   = (const float*)d_in[22];
  const float* br2   = (const float*)d_in[23];

  // workspace layout (~53 MB), 256B-aligned slices
  char* w = (char*)d_ws;
  auto take = [&](size_t bytes) {
    char* p = w;
    w += (bytes + 255) & ~(size_t)255;
    return p;
  };
  float*     h    = (float*)take((size_t)NN * DD * sizeof(float));
  float*     agg  = (float*)take((size_t)NN * DD * sizeof(float));
  float*     e    = (float*)take((size_t)NN * sizeof(float));
  float*     q    = (float*)take((size_t)BB * DD * sizeof(float));
  float*     qst  = (float*)take((size_t)BB * 2 * DD * sizeof(float));
  float*     hs   = (float*)take((size_t)3 * BB * DD * sizeof(float));
  float*     cs   = (float*)take((size_t)3 * BB * DD * sizeof(float));
  int*       segs = (int*)take((size_t)(BB + 1) * sizeof(int));
  _Float16*  wp   = (_Float16*)take((size_t)16 * 32 * 16 * sizeof(_Float16));

  hipMemsetAsync(qst, 0, (size_t)BB * 2 * DD * sizeof(float), stream);
  hipMemsetAsync(hs,  0, (size_t)3 * BB * DD * sizeof(float), stream);
  hipMemsetAsync(cs,  0, (size_t)3 * BB * DD * sizeof(float), stream);

  const int wavesN  = (NN + 15) / 16;              // 12500 tiles
  const int blocksN = (wavesN * 32 + 255) / 256;   // 8 waves / block

  k_pack_w<<<2, 256, 0, stream>>>(W0, Wm, Wih_g, Whh_g, wp);
  k_embed_lin0<<<blocksN, 256, 0, stream>>>(node_type, emb, wp, b0, h, NN);
  k_segstart<<<(NN + 255) / 256, 256, 0, stream>>>(gid, segs, NN, BB);

  for (int step = 0; step < 6; ++step) {
    hipMemsetAsync(agg, 0, (size_t)NN * DD * sizeof(float), stream);
    k_scatter<<<(int)(((long long)EE * 8 + 255) / 256), 256, 0, stream>>>(src, dst, h, agg, EE);
    k_gru<<<blocksN, 256, 0, stream>>>(agg, h, wp, bih_g, bhh_g, NN);
  }

  for (int step = 0; step < 6; ++step) {
    k_lstm_q<<<BB, 128, 0, stream>>>(qst, q, hs, cs,
                                     Wih0, Whh0, bih0, bhh0,
                                     Wih12, Whh12, bih12, bhh12);
    k_attn<<<BB, 256, 0, stream>>>(h, q, segs, e, qst);
  }

  k_readout<<<BB, 32, 0, stream>>>(qst, Wr1, br1, Wr2, br2, (float*)d_out);
}
